// PIPNet_36120674959616
// MI455X (gfx1250) — matrix operand order repaired and checked by reading, not compiled, not run on hardware
//
#include <hip/hip_runtime.h>
#include <hip/hip_bf16.h>

typedef __attribute__((ext_vector_type(16))) __bf16 v16bf;
typedef __attribute__((ext_vector_type(8)))  float  v8f;

#define ED 128
#define DD 64
#define NSLOTS (8*4*32*16)   // 16384 bf16 = 32 KB fragment blob

// ---------------------------------------------------------------------------
// Prep kernel: repack W1 [128x128] f32 (row n = output feature, col k = input)
// into bf16 WMMA B-fragment layout for V_WMMA_F32_16X16X32_BF16.
// B tile (t,c) is the 32x16 slice B[k][n] = W1[n][k], k in [32c,32c+32),
// n in [16t,16t+16). Per the ISA B layout: lanes 0-15 hold K=0..7,16..23
// (elements 0..15 -> K ascending), lanes 16-31 hold K=8..15,24..31.
// Slot s = ((t*4+c)*32 + lane)*16 + e.
// ---------------------------------------------------------------------------
__global__ void w1_repack_kernel(const float* __restrict__ W1,
                                 __bf16* __restrict__ w1frag) {
    int s = blockIdx.x * blockDim.x + threadIdx.x;
    if (s >= NSLOTS) return;
    int e    = s & 15;
    int l    = (s >> 4) & 31;
    int f    = s >> 9;          // 0..31 fragment id
    int c    = f & 3;           // K chunk
    int t    = f >> 2;          // N tile
    int n    = t * 16 + (l & 15);
    int k    = c * 32 + (e & 7) + 8 * (l >> 4) + 16 * (e >> 3);
    w1frag[s] = (__bf16)W1[n * ED + k];
}

// ---------------------------------------------------------------------------
// Fused: gather -> bf16 WMMA GEMM (x @ W1^T) -> +b1 -> ReLU -> dot(W2) -> +b2
// One wave = 16 pair-rows. 8 N-tiles x 4 K-steps = 32 WMMAs per wave.
// ---------------------------------------------------------------------------
__global__ __launch_bounds__(256)
void pipnet_fused_kernel(const float* __restrict__ g1, const float* __restrict__ g2,
                         const int* __restrict__ idxL, const int* __restrict__ idxR,
                         const float* __restrict__ b1, const float* __restrict__ w2,
                         const float* __restrict__ b2,
                         const __bf16* __restrict__ w1frag,
                         float* __restrict__ out, int P)
{
    __shared__ __align__(32) __bf16 ldsB[NSLOTS];   // 32 KB B fragments
    __shared__ float ldsBias[ED];
    __shared__ float ldsW2[ED];

    const int tid = threadIdx.x;

    // Stage W1 fragments into LDS: 32 KB linear copy, b128 granularity.
    {
        const uint4* src = (const uint4*)w1frag;
        uint4*       dst = (uint4*)ldsB;
        #pragma unroll
        for (int i = 0; i < 8; ++i)
            dst[tid + i * 256] = src[tid + i * 256];
    }
    if (tid < ED) { ldsBias[tid] = b1[tid]; ldsW2[tid] = w2[tid]; }
    __syncthreads();

    const int lane    = tid & 31;
    const int wave    = tid >> 5;
    const int m       = lane & 15;      // row within 16-row tile
    const int half    = lane >> 4;      // which K-half this lane carries
    const int rowBase = blockIdx.x * 128 + wave * 16;

    // Clamp gather row for the tail block: WMMA requires EXEC all-1s, so we
    // compute garbage rows on clamped data and only guard the final store.
    int row  = rowBase + m;
    int rowc = row < P ? row : P - 1;

    const float* srcL = g1 + (size_t)idxL[rowc] * DD;
    const float* srcR = g2 + (size_t)idxR[rowc] * DD;

    // Build 4 A fragments (16x32 bf16 each). For chunk c, lane needs
    // x[row][c*32 + 8*half + 0..7] and x[row][c*32 + 16 + 8*half + 0..7].
    // x = [left(64) | right(64)], so chunks 0,1 come from left, 2,3 from right.
    v16bf afrag[4];
    #pragma unroll
    for (int c = 0; c < 4; ++c) {
        const float* src = (c < 2) ? srcL : srcR;
        const int o = (c & 1) * 32 + 8 * half;
        float4 f0 = *(const float4*)(src + o);
        float4 f1 = *(const float4*)(src + o + 4);
        float4 f2 = *(const float4*)(src + o + 16);
        float4 f3 = *(const float4*)(src + o + 20);
        v16bf a;
        a[0]  = (__bf16)f0.x; a[1]  = (__bf16)f0.y; a[2]  = (__bf16)f0.z; a[3]  = (__bf16)f0.w;
        a[4]  = (__bf16)f1.x; a[5]  = (__bf16)f1.y; a[6]  = (__bf16)f1.z; a[7]  = (__bf16)f1.w;
        a[8]  = (__bf16)f2.x; a[9]  = (__bf16)f2.y; a[10] = (__bf16)f2.z; a[11] = (__bf16)f2.w;
        a[12] = (__bf16)f3.x; a[13] = (__bf16)f3.y; a[14] = (__bf16)f3.z; a[15] = (__bf16)f3.w;
        afrag[c] = a;
    }

    // h-tile epilogue note: D layout -> lane holds column n = tile*16 + m for
    // rows M = j + 8*half (j = VGPR index 0..7). So bias/W2 are per-lane
    // scalars per tile, and the W2 dot is a per-lane FMA into row partials.
    float partial[8] = {0.f, 0.f, 0.f, 0.f, 0.f, 0.f, 0.f, 0.f};

    #pragma unroll
    for (int t = 0; t < 8; ++t) {
        v8f acc = {};
        #pragma unroll
        for (int c = 0; c < 4; ++c) {
            const v16bf bfrag =
                *(const v16bf*)(ldsB + ((t * 4 + c) * 32 + lane) * 16);
            acc = __builtin_amdgcn_wmma_f32_16x16x32_bf16(
                      false, afrag[c], false, bfrag, (short)0, acc, false, false);
        }
        const int   n  = t * 16 + m;
        const float bb = ldsBias[n];
        const float ww = ldsW2[n];
        #pragma unroll
        for (int j = 0; j < 8; ++j) {
            float v = acc[j] + bb;
            v = v > 0.f ? v : 0.f;
            partial[j] += v * ww;
        }
    }

    // Reduce over the 16 lanes of each half (each lane held distinct n's).
    #pragma unroll
    for (int j = 0; j < 8; ++j) {
        partial[j] += __shfl_xor(partial[j], 8, 32);
        partial[j] += __shfl_xor(partial[j], 4, 32);
        partial[j] += __shfl_xor(partial[j], 2, 32);
        partial[j] += __shfl_xor(partial[j], 1, 32);
    }

    if (m == 0) {
        const float b2v = b2[0];
        #pragma unroll
        for (int j = 0; j < 8; ++j) {
            int r = rowBase + j + 8 * half;   // rows 0..7 from lane 0, 8..15 from lane 16
            if (r < P) out[r] = partial[j] + b2v;
        }
    }
}

extern "C" void kernel_launch(void* const* d_in, const int* in_sizes, int n_in,
                              void* d_out, int out_size, void* d_ws, size_t ws_size,
                              hipStream_t stream) {
    const float* g1   = (const float*)d_in[0];
    const float* g2   = (const float*)d_in[1];
    const int*   idxL = (const int*)d_in[2];
    const int*   idxR = (const int*)d_in[3];
    const float* W1   = (const float*)d_in[4];
    const float* b1   = (const float*)d_in[5];
    const float* w2   = (const float*)d_in[6];
    const float* b2   = (const float*)d_in[7];
    float*       out  = (float*)d_out;
    const int    P    = in_sizes[2];

    __bf16* w1frag = (__bf16*)d_ws;   // needs 32 KB of scratch

    w1_repack_kernel<<<NSLOTS / 256, 256, 0, stream>>>(W1, w1frag);

    const int blocks = (P + 127) / 128;
    pipnet_fused_kernel<<<blocks, 256, 0, stream>>>(
        g1, g2, idxL, idxR, b1, w2, b2, w1frag, out, P);
}
